// SparseMLP_34918084116583
// MI455X (gfx1250) — matrix operand order repaired and checked by reading, not compiled, not run on hardware
//
#include <hip/hip_runtime.h>

// ---------------------------------------------------------------------------
// Problem constants (match reference setup_inputs)
// ---------------------------------------------------------------------------
constexpr int B = 2, S = 2048, H = 2048, E = 8, I = 1408;
constexpr int T = B * S;          // 4096 tokens
constexpr int TB = 16;            // tokens per expert tile (WMMA M)

typedef __attribute__((ext_vector_type(16))) __bf16 v16bf;
typedef __attribute__((ext_vector_type(8)))  float  v8f;

union Frag16 {                    // 16 bf16 = one WMMA A/B operand (8 VGPRs)
    v16bf v;
    uint4 q[2];
};

#if __has_builtin(__builtin_amdgcn_global_load_async_to_lds_b128)
#define HAVE_ASYNC_LDS 1
// exact param types per clang diagnostic: (v4i __device__*, v4i __shared__*, int, int)
typedef int v4i __attribute__((vector_size(16)));
typedef __attribute__((address_space(1))) v4i* g_v4i_ptr;   // global (printed __device__)
typedef __attribute__((address_space(3))) v4i* l_v4i_ptr;   // LDS    (printed __shared__)
#else
#define HAVE_ASYNC_LDS 0
#endif

static __device__ __forceinline__ unsigned short bf16bits(float f) {
    unsigned int u = __float_as_uint(f);
    return (unsigned short)((u + 0x7FFFu + ((u >> 16) & 1u)) >> 16); // RNE
}
static __device__ __forceinline__ __bf16 f32_to_bf16(float f) {
    unsigned short s = bf16bits(f);
    return __builtin_bit_cast(__bf16, s);
}

// ---------------------------------------------------------------------------
// Kernel 1: tiled transpose f32[R][C] -> bf16[C][R], batched over gridDim.z
// (gate/up: [E][H][I] -> [E][I][H];  down: [E][I][H] -> [E][H][I])
// ---------------------------------------------------------------------------
__global__ __launch_bounds__(256) void transpose_to_bf16(
    const float* __restrict__ src, __bf16* __restrict__ dst, int R, int C)
{
    __shared__ float tile[32][33];
    const int e  = blockIdx.z;
    src += (size_t)e * R * C;
    dst += (size_t)e * R * C;
    const int c0 = blockIdx.x * 32, r0 = blockIdx.y * 32;
    const int tx = threadIdx.x & 31, tg = threadIdx.x >> 5;   // 8 row-groups

    #pragma unroll
    for (int i = 0; i < 32; i += 8) {
        int r = r0 + tg + i, c = c0 + tx;
        tile[tg + i][tx] = (r < R && c < C) ? src[(size_t)r * C + c] : 0.f;
    }
    __syncthreads();
    #pragma unroll
    for (int i = 0; i < 32; i += 8) {
        int c = c0 + tg + i, r = r0 + tx;
        if (c < C && r < R) dst[(size_t)c * R + r] = f32_to_bf16(tile[tx][tg + i]);
    }
}

// ---------------------------------------------------------------------------
// Kernel 2: x f32 -> bf16 (vectorized x4)
// ---------------------------------------------------------------------------
__global__ __launch_bounds__(256) void convert_x_kernel(
    const float* __restrict__ x, __bf16* __restrict__ o, int n4)
{
    int i = blockIdx.x * blockDim.x + threadIdx.x;
    if (i >= n4) return;
    float4 v = ((const float4*)x)[i];
    uint2 p;
    p.x = (unsigned int)bf16bits(v.x) | ((unsigned int)bf16bits(v.y) << 16);
    p.y = (unsigned int)bf16bits(v.z) | ((unsigned int)bf16bits(v.w) << 16);
    ((uint2*)o)[i] = p;
}

// ---------------------------------------------------------------------------
// Kernel 3: router.  One wave per token: 8 logits, softmax, top-2,
// write sparse scores and append token to per-expert lists.
// ---------------------------------------------------------------------------
__global__ __launch_bounds__(256) void router_kernel(
    const float* __restrict__ x, const float* __restrict__ wr,
    float* __restrict__ scores, int* __restrict__ cnt, int* __restrict__ toklist)
{
    const int wave = threadIdx.x >> 5;
    const int lane = threadIdx.x & 31;
    const int t = blockIdx.x * 8 + wave;
    if (t >= T) return;

    float acc[E] = {};
    const float* xr = x + (size_t)t * H;
    for (int h = lane; h < H; h += 32) {
        float xv = xr[h];
        const float* w = wr + (size_t)h * E;
        #pragma unroll
        for (int e = 0; e < E; ++e) acc[e] = fmaf(xv, w[e], acc[e]);
    }
    #pragma unroll
    for (int e = 0; e < E; ++e)
        for (int off = 16; off; off >>= 1) acc[e] += __shfl_xor(acc[e], off, 32);

    if (lane == 0) {
        float mx = acc[0];
        #pragma unroll
        for (int e = 1; e < E; ++e) mx = fmaxf(mx, acc[e]);
        float p[E], sum = 0.f;
        #pragma unroll
        for (int e = 0; e < E; ++e) { p[e] = __expf(acc[e] - mx); sum += p[e]; }
        float inv = 1.0f / sum;

        int i1 = 0; float b1 = p[0];
        #pragma unroll
        for (int e = 1; e < E; ++e) if (p[e] > b1) { b1 = p[e]; i1 = e; }
        int i2 = (i1 == 0) ? 1 : 0; float b2 = p[i2];
        #pragma unroll
        for (int e = 0; e < E; ++e)
            if (e != i1 && p[e] > b2) { b2 = p[e]; i2 = e; }

        #pragma unroll
        for (int e = 0; e < E; ++e)
            scores[(size_t)t * E + e] = (e == i1 || e == i2) ? p[e] * inv : 0.f;

        int s1 = atomicAdd(&cnt[i1], 1); toklist[i1 * T + s1] = t;
        int s2 = atomicAdd(&cnt[i2], 1); toklist[i2 * T + s2] = t;
    }
}

// ---------------------------------------------------------------------------
// Kernel 4: fused sparse expert MLP.  One WG = 16 tokens x 1 expert.
//   phase 1: act = silu(x Wg) * (x Wu)      (K = H = 2048, N = I = 1408)
//   phase 2: out += score * (act Wd)        (K = I = 1408, N = H = 2048)
// A fragments from LDS (conflict-free b128), B fragments are contiguous
// 32-byte per-lane loads from the pre-transposed bf16 weights (L2-resident).
// ---------------------------------------------------------------------------
constexpr int XS_STRIDE  = H + 8;   // halves; +16B pad -> conflict-free banks
constexpr int ACT_STRIDE = I + 8;

__global__ __launch_bounds__(128) void expert_kernel(
    const __bf16* __restrict__ xbf,  const __bf16* __restrict__ wg_t,
    const __bf16* __restrict__ wu_t, const __bf16* __restrict__ wd_t,
    const float*  __restrict__ scores, const int* __restrict__ cnt,
    const int*    __restrict__ toklist, float* __restrict__ out)
{
    const int e    = blockIdx.y;
    const int tile = blockIdx.x;
    const int nTok = cnt[e];
    if (tile * TB >= nTok) return;

    __shared__ __bf16 xs[TB * XS_STRIDE];     // 64 KB  x tile (bf16)
    __shared__ __bf16 acts[TB * ACT_STRIDE];  // 45 KB  activation tile
    __shared__ int    tokS[TB];
    __shared__ float  scrS[TB];

    const int tid = threadIdx.x;
    if (tid < TB) {
        int slot = tile * TB + tid;
        int t = (slot < nTok) ? toklist[e * T + slot] : -1;
        tokS[tid] = t;
        scrS[tid] = (t >= 0) ? scores[(size_t)t * E + e] : 0.f;
    }
    __syncthreads();

    // ---- stage 16 x-rows (2048 bf16 each) into LDS, 128 threads ----
    {
        const int m = tid >> 3, l8 = tid & 7;   // 8 threads per row
        int t = tokS[m]; if (t < 0) t = 0;
        const uint4* src = (const uint4*)(xbf + (size_t)t * H);  // 256 uint4/row
        uint4* dst = (uint4*)(xs + m * XS_STRIDE);
#if HAVE_ASYNC_LDS
        // CDNA5: direct global->LDS async copies (ASYNCcnt), no VGPR round-trip
        #pragma unroll
        for (int c = 0; c < 32; ++c) {
            int idx = c * 8 + l8;
            __builtin_amdgcn_global_load_async_to_lds_b128(
                (g_v4i_ptr)(src + idx), (l_v4i_ptr)(dst + idx), 0, 0);
        }
        asm volatile("s_wait_asynccnt 0x0" ::: "memory");
#else
        #pragma unroll
        for (int c = 0; c < 32; ++c) {
            int idx = c * 8 + l8;
            dst[idx] = src[idx];
        }
#endif
    }
    __syncthreads();

    const int lane = tid & 31, wave = tid >> 5;
    const int lp = lane & 15, hf = lane >> 4;

    // ---- phase 1: gate/up GEMMs + SiLU, N strip-mined over 4 waves ----
    const __bf16* xrow = xs + lp * XS_STRIDE;       // A: token row lp
    for (int i0 = wave * 16; i0 < I; i0 += 64) {
        v8f accg = {}, accu = {};
        const __bf16* bg = wg_t + ((size_t)e * I + (i0 + lp)) * H;  // B col i0+lp
        const __bf16* bu = wu_t + ((size_t)e * I + (i0 + lp)) * H;
        for (int k0 = 0; k0 < H; k0 += 32) {
            Frag16 a, fg, fu;
            a.q[0]  = *(const uint4*)(xrow + k0 + 16 * hf);
            a.q[1]  = *(const uint4*)(xrow + k0 + 16 * hf + 8);
            const uint4* gp = (const uint4*)(bg + k0 + 16 * hf);
            fg.q[0] = gp[0]; fg.q[1] = gp[1];
            const uint4* up = (const uint4*)(bu + k0 + 16 * hf);
            fu.q[0] = up[0]; fu.q[1] = up[1];
            // branch-free prefetch: H is a power of two, wrap in-bounds
            const int kpf = (k0 + 256) & (H - 1);
            __builtin_prefetch(bg + kpf, 0, 1);
            __builtin_prefetch(bu + kpf, 0, 1);
            accg = __builtin_amdgcn_wmma_f32_16x16x32_bf16(
                       false, a.v, false, fg.v, (short)0, accg, false, false);
            accu = __builtin_amdgcn_wmma_f32_16x16x32_bf16(
                       false, a.v, false, fu.v, (short)0, accu, false, false);
        }
        #pragma unroll
        for (int r = 0; r < 8; ++r) {                // C layout: m = r + 8*hf, n = lp
            float g = accg[r], u = accu[r];
            float sv = g * __builtin_amdgcn_rcpf(1.0f + __expf(-g)) * u;
            acts[(r + 8 * hf) * ACT_STRIDE + i0 + lp] = f32_to_bf16(sv);
        }
    }
    __syncthreads();

    // ---- phase 2: down GEMM, score-weighted accumulate into out ----
    const __bf16* arow = acts + lp * ACT_STRIDE;
    for (int h0 = wave * 16; h0 < H; h0 += 64) {
        v8f acc = {};
        const __bf16* bd = wd_t + ((size_t)e * H + (h0 + lp)) * I;
        for (int k0 = 0; k0 < I; k0 += 32) {
            Frag16 a, fb;
            a.q[0]  = *(const uint4*)(arow + k0 + 16 * hf);
            a.q[1]  = *(const uint4*)(arow + k0 + 16 * hf + 8);
            const uint4* bp = (const uint4*)(bd + k0 + 16 * hf);
            fb.q[0] = bp[0]; fb.q[1] = bp[1];
            // branch-free prefetch: clamp via select (I not a power of two)
            int kpf = k0 + 256; kpf = (kpf < I) ? kpf : 0;
            __builtin_prefetch(bd + kpf, 0, 1);
            acc = __builtin_amdgcn_wmma_f32_16x16x32_bf16(
                      false, a.v, false, fb.v, (short)0, acc, false, false);
        }
        #pragma unroll
        for (int r = 0; r < 8; ++r) {
            int m = r + 8 * hf;
            int t = tokS[m];
            if (t >= 0)
                atomicAdd(&out[(size_t)t * H + h0 + lp], scrS[m] * acc[r]);
        }
    }
}

// ---------------------------------------------------------------------------
// Host-side launcher
// ---------------------------------------------------------------------------
extern "C" void kernel_launch(void* const* d_in, const int* in_sizes, int n_in,
                              void* d_out, int out_size, void* d_ws, size_t ws_size,
                              hipStream_t stream)
{
    const float* x  = (const float*)d_in[0];   // [B,S,H]
    const float* wr = (const float*)d_in[1];   // [H,E]
    const float* wg = (const float*)d_in[2];   // [E,H,I]
    const float* wu = (const float*)d_in[3];   // [E,H,I]
    const float* wd = (const float*)d_in[4];   // [E,I,H]
    float* out = (float*)d_out;                // [T*H] out ++ [T*E] scores

    // workspace layout
    const size_t W_ELEMS = (size_t)E * H * I;          // 23,068,672
    char* w = (char*)d_ws;
    __bf16* wg_t = (__bf16*)w;            w += W_ELEMS * sizeof(__bf16);
    __bf16* wu_t = (__bf16*)w;            w += W_ELEMS * sizeof(__bf16);
    __bf16* wd_t = (__bf16*)w;            w += W_ELEMS * sizeof(__bf16);
    __bf16* xbf  = (__bf16*)w;            w += (size_t)T * H * sizeof(__bf16);
    int*    cnt  = (int*)w;               w += 16 * sizeof(int);   // padded
    int*    tokl = (int*)w;               /* E*T ints */

    float* scores = out + (size_t)T * H;

    // zero the accumulated output and the expert counters
    (void)hipMemsetAsync(out, 0, (size_t)T * H * sizeof(float), stream);
    (void)hipMemsetAsync(cnt, 0, 16 * sizeof(int), stream);

    // 1) weights -> bf16, transposed for contiguous-K B-fragment loads
    transpose_to_bf16<<<dim3(I / 32, H / 32, E), 256, 0, stream>>>(wg, wg_t, H, I);
    transpose_to_bf16<<<dim3(I / 32, H / 32, E), 256, 0, stream>>>(wu, wu_t, H, I);
    transpose_to_bf16<<<dim3(H / 32, I / 32, E), 256, 0, stream>>>(wd, wd_t, I, H);

    // 2) activations -> bf16
    const int n4 = T * H / 4;
    convert_x_kernel<<<(n4 + 255) / 256, 256, 0, stream>>>(x, xbf, n4);

    // 3) router: scores + per-expert token lists
    router_kernel<<<T / 8, 256, 0, stream>>>(x, wr, scores, cnt, tokl);

    // 4) sparse expert MLP (tiles past each expert's count exit immediately)
    expert_kernel<<<dim3(T / TB, E), 128, 0, stream>>>(
        xbf, wg_t, wu_t, wd_t, scores, cnt, tokl, out);
}
